// CrossStateFusion_88940182766310
// MI455X (gfx1250) — matrix operand order repaired and checked by reading, not compiled, not run on hardware
//
#include <hip/hip_runtime.h>
#include <math.h>

// ---------------------------------------------------------------------------
// CrossStateFusion for MI455X (gfx1250, wave32, WMMA).
//
// Roofline (corrected): ~7.3e10 FLOP after algebraic pooling vs ~140 MB HBM
// => memory floor ~6us @ 23.3 TB/s. Matrix-op ceilings by K-scaling from the
// fp8 spec (20.1 PTOPS sparse -> ~10 PFLOPS dense, K=128):
//   f32 WMMA (K=4)  ~313 TFLOPS -> 233us  (compute-bound, wrong choice)
//   f16 WMMA (K=32) ~2.5 PFLOPS ->  29us  (chosen: f16 in, f32 accumulate)
//   fp8 (K=128)     ~10  PFLOPS -> 7.3us  (hits roofline but too lossy vs f32)
// So: v_wmma_f32_16x16x32_f16 for every GEMM; softmax/LN/poolings stay f32.
// Weights are converted to f16 once (4 MB ws slab), amortized over 512 WGs.
// ---------------------------------------------------------------------------

typedef __attribute__((ext_vector_type(2)))  float    v2f;
typedef __attribute__((ext_vector_type(8)))  float    v8f;
typedef __attribute__((ext_vector_type(8)))  _Float16 v8h;
typedef __attribute__((ext_vector_type(16))) _Float16 v16h;

#define B_   8
#define K_   4
#define L_   1024
#define D_   512
#define H_   8
#define HD_  64
#define EPS_    1e-8f
#define LN_EPS_ 1e-5f
#define SCALE_  0.125f  /* 64^-0.5 */

// d_out flat offsets (floats), outputs concatenated in return order
#define FUSED_OFF 0
#define AP_OFF    4194304
#define WM_OFF    8388608
#define MF_OFF    12582912
#define AM_OFF    16777216
#define NSW_OFF   16809984

// ws layout: [0..64) floats nsw+logw; then f16 weight slab (32B aligned)
#define HW_Q   0
#define HW_K   262144
#define HW_V   524288
#define HW_O   786432
#define HW_G1  1048576   /* 512x1536 */
#define HW_G2  1835008   /* 512x512  */
#define HW_TOT 2097152

// --- f16 WMMA fragment helpers (D = A(16x32) * B(32x16) + C, f32 acc) ------
// A layout (16-bit A 16x32): lane l: M=l%16; component j -> K=j+16*hl
//   => two contiguous 8-half (16B) chunks (matches B's k order).
// B layout (y = X*W^T => B[k][n] = W[n][k]): component j -> K=j+16*hl
//   => one contiguous 16-half (32B) chunk of W row n0+l%16.
// C/D: lane l, vgpr i -> D[i+8*hl][l%16].
static __device__ __forceinline__ v16h afrag(const _Float16* __restrict__ row,
                                             int kk, int hl) {
  v8h a0 = *(const v8h*)(row + kk + 16 * hl);
  v8h a1 = *(const v8h*)(row + kk + 16 * hl + 8);
  return __builtin_shufflevector(a0, a1, 0, 1, 2, 3, 4, 5, 6, 7, 8, 9, 10, 11,
                                 12, 13, 14, 15);
}
static __device__ __forceinline__ v16h bfrag(const _Float16* __restrict__ row,
                                             int kk, int hl) {
  return *(const v16h*)(row + kk + 16 * hl);
}
static __device__ __forceinline__ v8f wmma16(v16h a, v16h b, v8f c) {
  return __builtin_amdgcn_wmma_f32_16x16x32_f16(false, a, false, b, (short)0,
                                                c, false, false);
}

// ---------------------------------------------------------------------------
// k_cvt: f32 -> f16 weight conversion (n multiple of 2048)
// ---------------------------------------------------------------------------
__global__ __launch_bounds__(256) void k_cvt(const float* __restrict__ src,
                                             _Float16* __restrict__ dst,
                                             int n) {
  int i = (blockIdx.x * 256 + threadIdx.x) * 8;
  if (i + 8 <= n)
    for (int j = 0; j < 8; ++j) dst[i + j] = (_Float16)src[i + j];
}

// ---------------------------------------------------------------------------
// k0: normalized state weights nsw[b][k] + log(max(nsw,EPS)) into workspace
// ---------------------------------------------------------------------------
__global__ void k0_nsw(const float* __restrict__ sw,
                       const unsigned char* __restrict__ pres,
                       float* __restrict__ out, float* __restrict__ ws) {
  int i = threadIdx.x;
  if (i >= B_ * K_) return;
  int b = i / K_;
  float denom = 0.f, psum = 0.f, wk[K_], pk[K_];
  for (int k = 0; k < K_; ++k) {
    pk[k] = pres[b * K_ + k] ? 1.f : 0.f;
    wk[k] = sw[b * K_ + k] * pk[k];
    denom += wk[k];
    psum += pk[k];
  }
  int k = i % K_;
  float fb = pk[k] / fmaxf(psum, 1.f);
  float nsw = (denom > EPS_) ? (wk[k] / fmaxf(denom, EPS_)) : fb;
  out[NSW_OFF + i] = nsw;
  ws[i] = nsw;
  ws[32 + i] = logf(fmaxf(nsw, EPS_));
}

// ---------------------------------------------------------------------------
// k1: weighted_mean + max_feat (streams state_repr once; pure f32)
// ---------------------------------------------------------------------------
__global__ __launch_bounds__(256) void k1_pool(
    const float* __restrict__ sr, const unsigned char* __restrict__ rmask,
    const int* __restrict__ roles, const unsigned char* __restrict__ pres,
    const float* __restrict__ remb, const float* __restrict__ ws,
    float* __restrict__ out) {
  int t = blockIdx.x;
  int b = t / L_, l = t % L_;
  float swr[K_];
  int rol[K_], vmask[K_], anyv = 0;
  float s = 0.f;
  for (int k = 0; k < K_; ++k) {
    int v = (rmask[(b * K_ + k) * L_ + l] != 0) && (pres[b * K_ + k] != 0);
    vmask[k] = v;
    anyv |= v;
    rol[k] = max(roles[b * K_ + k], 0);
    swr[k] = ws[b * K_ + k] * (v ? 1.f : 0.f);
    s += swr[k];
  }
  float inv = 1.f / fmaxf(s, EPS_);
  for (int k = 0; k < K_; ++k) swr[k] *= inv;

  int d = threadIdx.x * 2;
  v2f wm; wm.x = 0.f; wm.y = 0.f;
  v2f mf; mf.x = -1e9f; mf.y = -1e9f;
  for (int k = 0; k < K_; ++k) {
    v2f x; x.x = 0.f; x.y = 0.f;
    if (vmask[k]) {
      v2f a = *(const v2f*)(sr + ((size_t)(b * K_ + k) * L_ + l) * D_ + d);
      v2f e = *(const v2f*)(remb + rol[k] * D_ + d);
      x.x = a.x + e.x; x.y = a.y + e.y;
      mf.x = fmaxf(mf.x, x.x); mf.y = fmaxf(mf.y, x.y);
    }
    wm.x += x.x * swr[k]; wm.y += x.y * swr[k];
  }
  if (!anyv) { mf.x = 0.f; mf.y = 0.f; }
  *(v2f*)(out + WM_OFF + (size_t)t * D_ + d) = wm;
  *(v2f*)(out + MF_OFF + (size_t)t * D_ + d) = mf;
}

// ---------------------------------------------------------------------------
// k2: fused QK projection + 4-key attention + pooled V/Wo projection.
// One workgroup = 16 tokens (64 x-rows). Algebra: pooling commutes with the
// linear V/Wo maps =>  attn_pooled = [sum_q swr*attn-weighted x] @ Wv_h^T
// per head, concatenated, then @ Wo^T at M=16.  Dynamic LDS ~131KB.
// ---------------------------------------------------------------------------
#define XLDH 520  /* halves: 512+8, keeps 16B align; rows 4 banks apart */
#define QLD  66   /* f32 Q/K scratch stride */

__global__ __launch_bounds__(256) void k2_attn(
    const float* __restrict__ sr, const unsigned char* __restrict__ rmask,
    const int* __restrict__ roles, const unsigned char* __restrict__ pres,
    const float* __restrict__ remb, const float* __restrict__ ws,
    const _Float16* __restrict__ hw, float* __restrict__ out) {
  extern __shared__ char smraw[];
  _Float16* Xh = (_Float16*)smraw;                       // 64 x XLDH
  float* Qs = (float*)(smraw + 64 * XLDH * 2);           // 64 x QLD
  float* Ks = Qs + 64 * QLD;                             // 64 x QLD
  _Float16* xvh = (_Float16*)(Ks + 64 * QLD);            // 16 x XLDH
  _Float16* Ph  = xvh + 16 * XLDH;                       // 16 x XLDH
  __shared__ float As[16 * 16];   // attn[lt][q*4+kk]
  __shared__ float wA[64];        // sum_q swr[q]*attn[q][kk]
  __shared__ float am[64];        // attn_mean accumulator
  __shared__ float swr_s[64];
  __shared__ float logw_s[4];
  __shared__ int role_s[4];
  __shared__ int valid_s[64];

  const _Float16* hWq = hw + HW_Q;
  const _Float16* hWk = hw + HW_K;
  const _Float16* hWv = hw + HW_V;
  const _Float16* hWo = hw + HW_O;

  const int tid = threadIdx.x;
  const int lane = tid & 31, wv = tid >> 5;
  const int hl = lane >> 4, ln = lane & 15;
  const int t0 = blockIdx.x * 16;
  const int b = t0 / L_;  // 16 | L, constant per block
  const int l0 = t0 % L_;

  if (tid < 4) {
    logw_s[tid] = ws[32 + b * K_ + tid];
    role_s[tid] = max(roles[b * K_ + tid], 0);
  }
  if (tid < 64) {
    int lt = tid >> 2, k = tid & 3;
    int v = (rmask[(b * K_ + k) * L_ + l0 + lt] != 0) &&
            (pres[b * K_ + k] != 0);
    valid_s[tid] = v;
    swr_s[tid] = ws[b * K_ + k] * (v ? 1.f : 0.f);
    am[tid] = 0.f;
  }
  __syncthreads();
  if (tid < 16) {
    float ss = swr_s[tid * 4] + swr_s[tid * 4 + 1] + swr_s[tid * 4 + 2] +
               swr_s[tid * 4 + 3];
    float inv = 1.f / fmaxf(ss, EPS_);
    for (int k = 0; k < 4; ++k) swr_s[tid * 4 + k] *= inv;
  }
  // Stage masked x rows (f16) into LDS: row r = lt*4+k -> x[b, l0+lt, k, :]
  {
    int d = tid * 2;
    for (int r = 0; r < 64; ++r) {
      int k = r & 3, lt = r >> 2;
      float x0 = 0.f, x1 = 0.f;
      if (valid_s[r]) {
        v2f a = *(const v2f*)(sr + ((size_t)(b * K_ + k) * L_ + l0 + lt) * D_ + d);
        v2f e = *(const v2f*)(remb + role_s[k] * D_ + d);
        x0 = a.x + e.x; x1 = a.y + e.y;
      }
      Xh[r * XLDH + d] = (_Float16)x0;
      Xh[r * XLDH + d + 1] = (_Float16)x1;
    }
  }
  __syncthreads();

  for (int h = 0; h < H_; ++h) {
    // Warm next head's f16 weight rows in cache (global_prefetch_b8)
    if (h + 1 < H_) {
      size_t rr = (size_t)((h + 1) * HD_ + (tid & 63)) * D_ + (tid >> 6) * 128;
      __builtin_prefetch(hWq + rr, 0, 0);
      __builtin_prefetch(hWk + rr, 0, 0);
      __builtin_prefetch(hWv + rr, 0, 0);
    }
    // Qh/Kh = X @ W_h^T : 2 * (4x4 tiles of 16x16) over 8 waves = 4 each
    for (int it = 0; it < 4; ++it) {
      int g = wv * 4 + it;
      int mat = g >> 4, tl = g & 15, mt = tl >> 2, nt = tl & 3;
      const _Float16* W = mat ? hWk : hWq;
      const _Float16* wrow = W + (size_t)(h * HD_ + nt * 16 + ln) * D_;
      const _Float16* arow = Xh + (mt * 16 + ln) * XLDH;
      v8f c = {0.f, 0.f, 0.f, 0.f, 0.f, 0.f, 0.f, 0.f};
      for (int kk = 0; kk < D_; kk += 32)
        c = wmma16(afrag(arow, kk, hl), bfrag(wrow, kk, hl), c);
      float* Dst = mat ? Ks : Qs;
      for (int i = 0; i < 8; ++i)
        Dst[(mt * 16 + i + 8 * hl) * QLD + nt * 16 + ln] = c[i];
    }
    __syncthreads();
    // logits + masked softmax: one lane per (token, q-state); f32
    if (tid < 64) {
      int lt = tid >> 2, q = tid & 3;
      const float* qr = Qs + (lt * 4 + q) * QLD;
      float lg[4], m = -3.0e38f;
      int any = 0;
      for (int kk = 0; kk < 4; ++kk) {
        const float* kr = Ks + (lt * 4 + kk) * QLD;
        float dot = 0.f;
        for (int d = 0; d < HD_; ++d) dot += qr[d] * kr[d];
        lg[kk] = dot * SCALE_ + logw_s[kk];
        if (valid_s[lt * 4 + kk]) { any = 1; m = fmaxf(m, lg[kk]); }
      }
      float e[4], ssum = 0.f;
      for (int kk = 0; kk < 4; ++kk) {
        e[kk] = (any && valid_s[lt * 4 + kk]) ? expf(lg[kk] - m) : 0.f;
        ssum += e[kk];
      }
      float inv = any ? (1.f / ssum) : 0.f;  // nan_to_num(all-masked) -> 0
      for (int kk = 0; kk < 4; ++kk) As[lt * 16 + q * 4 + kk] = e[kk] * inv;
    }
    __syncthreads();
    // attn_mean accum + swr-weighted attn coefficients
    if (tid < 64) {
      int lt = tid >> 2, kk = tid & 3;
      float s = 0.f, wa = 0.f;
      for (int q = 0; q < 4; ++q) {
        float a = As[lt * 16 + q * 4 + kk];
        s += a;
        wa += swr_s[lt * 4 + q] * a;
      }
      am[tid] += s;
      wA[tid] = wa;
    }
    __syncthreads();
    // xv[lt][d] = sum_kk wA[lt][kk] * x[lt,kk][d]   (pool before Wv!)
    for (int idx = tid; idx < 16 * D_; idx += 256) {
      int lt = idx >> 9, d = idx & 511;
      float p = 0.f;
      for (int kk = 0; kk < 4; ++kk)
        p += wA[lt * 4 + kk] * (float)Xh[(lt * 4 + kk) * XLDH + d];
      xvh[lt * XLDH + d] = (_Float16)p;
    }
    __syncthreads();
    // P[:, h*64 + n] = xv @ Wv_h^T : M=16, 4 N-tiles on waves 0..3
    if (wv < 4) {
      int n0 = wv * 16;
      const _Float16* wrow = hWv + (size_t)(h * HD_ + n0 + ln) * D_;
      const _Float16* arow = xvh + ln * XLDH;
      v8f c = {0.f, 0.f, 0.f, 0.f, 0.f, 0.f, 0.f, 0.f};
      for (int kk = 0; kk < D_; kk += 32)
        c = wmma16(afrag(arow, kk, hl), bfrag(wrow, kk, hl), c);
      for (int i = 0; i < 8; ++i)
        Ph[(i + 8 * hl) * XLDH + h * HD_ + n0 + ln] = (_Float16)c[i];
    }
    __syncthreads();
  }

  if (tid < 64) {
    int lt = tid >> 2, kk = tid & 3;
    out[AM_OFF + (size_t)(t0 + lt) * 4 + kk] = am[tid] * (1.f / 32.f);
  }
  // attn_pooled = P (16x512) @ Wo^T : 32 N-tiles over 8 waves
  for (int j = 0; j < 4; ++j) {
    int n0 = (wv * 4 + j) * 16;
    const _Float16* wrow = hWo + (size_t)(n0 + ln) * D_;
    const _Float16* arow = Ph + ln * XLDH;
    v8f c = {0.f, 0.f, 0.f, 0.f, 0.f, 0.f, 0.f, 0.f};
    for (int kk = 0; kk < D_; kk += 32)
      c = wmma16(afrag(arow, kk, hl), bfrag(wrow, kk, hl), c);
    for (int i = 0; i < 8; ++i)
      out[AP_OFF + (size_t)(t0 + i + 8 * hl) * D_ + n0 + ln] = c[i];
  }
}

// ---------------------------------------------------------------------------
// k3: gated fusion MLP per 16 tokens. LN in f32, GEMMs in f16 WMMA.
// ---------------------------------------------------------------------------
#define GLD  1540  /* f32 raw gate_in stride */
#define GLDH 1544  /* f16 LN'ed gate_in stride (16B-aligned chunks) */
#define HLDH 520   /* f16 hidden stride */
#define FLD  516   /* f32 fused stride */

__global__ __launch_bounds__(256) void k3_gate(
    const float* __restrict__ bg1, const float* __restrict__ bg2,
    const float* __restrict__ ln1g, const float* __restrict__ ln1b,
    const float* __restrict__ ng, const float* __restrict__ nb,
    const _Float16* __restrict__ hw, float* __restrict__ out) {
  extern __shared__ char smraw[];
  float* Gs = (float*)smraw;                              // 16 x GLD  raw
  _Float16* Gh = (_Float16*)(smraw + 16 * GLD * 4);       // 16 x GLDH LN'ed
  _Float16* Hs = Gh + 16 * GLDH;                          // 16 x HLDH gelu
  float* Fs = (float*)(Hs + 16 * HLDH);                   // 16 x FLD  fused
  __shared__ float red1[256], red2[256];
  __shared__ float mu_s[16], rs_s[16];

  const _Float16* hWg1 = hw + HW_G1;
  const _Float16* hWg2 = hw + HW_G2;

  const int tid = threadIdx.x;
  const int lane = tid & 31, wv = tid >> 5;
  const int hl = lane >> 4, ln = lane & 15;
  const int t0 = blockIdx.x * 16;

  for (int idx = tid; idx < 16 * 1536; idx += 256) {
    int r = idx / 1536, c = idx % 1536;
    size_t off = (c < 512) ? AP_OFF : (c < 1024) ? (size_t)WM_OFF : (size_t)MF_OFF;
    Gs[r * GLD + c] = out[off + (size_t)(t0 + r) * D_ + (c & 511)];
  }
  __syncthreads();
  {  // LN1 stats over 1536 (f32)
    int r = tid >> 4, seg = tid & 15;
    float s1 = 0.f, s2 = 0.f;
    for (int c = seg * 96; c < seg * 96 + 96; ++c) {
      float v = Gs[r * GLD + c];
      s1 += v; s2 += v * v;
    }
    red1[tid] = s1; red2[tid] = s2;
  }
  __syncthreads();
  if (tid < 16) {
    float s1 = 0.f, s2 = 0.f;
    for (int j = 0; j < 16; ++j) { s1 += red1[tid * 16 + j]; s2 += red2[tid * 16 + j]; }
    float mu = s1 / 1536.f;
    float var = s2 / 1536.f - mu * mu;
    mu_s[tid] = mu;
    rs_s[tid] = rsqrtf(var + LN_EPS_);
  }
  __syncthreads();
  // Gh = f16(LN(gate_in))
  for (int idx = tid; idx < 16 * 1536; idx += 256) {
    int r = idx / 1536, c = idx % 1536;
    float v = (Gs[r * GLD + c] - mu_s[r]) * rs_s[r] * ln1g[c] + ln1b[c];
    Gh[r * GLDH + c] = (_Float16)v;
  }
  __syncthreads();
  // h = gelu(Gh @ Wg1^T + bg1) : M=16, N=512, K=1536
  for (int j = 0; j < 4; ++j) {
    int n0 = (wv * 4 + j) * 16;
    const _Float16* wrow = hWg1 + (size_t)(n0 + ln) * 1536;
    const _Float16* arow = Gh + ln * GLDH;
    v8f c = {0.f, 0.f, 0.f, 0.f, 0.f, 0.f, 0.f, 0.f};
    for (int kk = 0; kk < 1536; kk += 32)
      c = wmma16(afrag(arow, kk, hl), bfrag(wrow, kk, hl), c);
    for (int i = 0; i < 8; ++i) {
      int r = i + 8 * hl, n = n0 + ln;
      float hv = c[i] + bg1[n];
      float ge = 0.5f * hv * (1.f + erff(hv * 0.70710678118654752f));  // exact
      Hs[r * HLDH + n] = (_Float16)ge;
    }
  }
  __syncthreads();
  // gate = sigmoid(h @ Wg2^T + bg2); fuse with raw ap/wm/mf (f32)
  for (int j = 0; j < 4; ++j) {
    int n0 = (wv * 4 + j) * 16;
    const _Float16* wrow = hWg2 + (size_t)(n0 + ln) * D_;
    const _Float16* arow = Hs + ln * HLDH;
    v8f c = {0.f, 0.f, 0.f, 0.f, 0.f, 0.f, 0.f, 0.f};
    for (int kk = 0; kk < D_; kk += 32)
      c = wmma16(afrag(arow, kk, hl), bfrag(wrow, kk, hl), c);
    for (int i = 0; i < 8; ++i) {
      int r = i + 8 * hl, n = n0 + ln;
      float z = c[i] + bg2[n];
      float gate = 1.f / (1.f + expf(-z));
      float ap = Gs[r * GLD + n];
      float wm = Gs[r * GLD + 512 + n];
      float mf = Gs[r * GLD + 1024 + n];
      float fb = 0.5f * (ap + wm);
      Fs[r * FLD + n] = gate * fb + (1.f - gate) * mf + wm;
    }
  }
  __syncthreads();
  {  // LN2 stats over 512 (f32)
    int r = tid >> 4, seg = tid & 15;
    float s1 = 0.f, s2 = 0.f;
    for (int c = seg * 32; c < seg * 32 + 32; ++c) {
      float v = Fs[r * FLD + c];
      s1 += v; s2 += v * v;
    }
    red1[tid] = s1; red2[tid] = s2;
  }
  __syncthreads();
  if (tid < 16) {
    float s1 = 0.f, s2 = 0.f;
    for (int j = 0; j < 16; ++j) { s1 += red1[tid * 16 + j]; s2 += red2[tid * 16 + j]; }
    float mu = s1 / 512.f;
    float var = s2 / 512.f - mu * mu;
    mu_s[tid] = mu;
    rs_s[tid] = rsqrtf(var + LN_EPS_);
  }
  __syncthreads();
  for (int idx = tid; idx < 16 * 512; idx += 256) {
    int r = idx >> 9, cc = idx & 511;
    float v = (Fs[r * FLD + cc] - mu_s[r]) * rs_s[r] * ng[cc] + nb[cc];
    out[FUSED_OFF + (size_t)(t0 + r) * D_ + cc] = v;
  }
}

// ---------------------------------------------------------------------------
extern "C" void kernel_launch(void* const* d_in, const int* in_sizes, int n_in,
                              void* d_out, int out_size, void* d_ws,
                              size_t ws_size, hipStream_t stream) {
  const float* sr = (const float*)d_in[0];
  const unsigned char* rmask = (const unsigned char*)d_in[1];
  const float* sw = (const float*)d_in[2];
  const int* roles = (const int*)d_in[3];
  const unsigned char* pres = (const unsigned char*)d_in[4];
  const float* remb = (const float*)d_in[5];
  const float* Wq = (const float*)d_in[6];
  const float* Wk = (const float*)d_in[7];
  const float* Wv = (const float*)d_in[8];
  const float* Wo = (const float*)d_in[9];
  const float* ln1g = (const float*)d_in[10];
  const float* ln1b = (const float*)d_in[11];
  const float* Wg1 = (const float*)d_in[12];
  const float* bg1 = (const float*)d_in[13];
  const float* Wg2 = (const float*)d_in[14];
  const float* bg2 = (const float*)d_in[15];
  const float* ng = (const float*)d_in[16];
  const float* nb = (const float*)d_in[17];
  float* out = (float*)d_out;
  float* ws = (float*)d_ws;
  _Float16* hw = (_Float16*)(ws + 64);  // f16 weight slab

  size_t sm2 = (size_t)64 * XLDH * 2 + (size_t)2 * 64 * QLD * 4 +
               (size_t)32 * XLDH * 2;                       // ~131 KB
  size_t sm3 = (size_t)16 * GLD * 4 + (size_t)16 * GLDH * 2 +
               (size_t)16 * HLDH * 2 + (size_t)16 * FLD * 4;  // ~193 KB
  (void)hipFuncSetAttribute((const void*)k2_attn,
                            hipFuncAttributeMaxDynamicSharedMemorySize,
                            (int)sm2);
  (void)hipFuncSetAttribute((const void*)k3_gate,
                            hipFuncAttributeMaxDynamicSharedMemorySize,
                            (int)sm3);

  // weight f32 -> f16 conversion (one-time per launch, 4 MB)
  k_cvt<<<262144 / 2048, 256, 0, stream>>>(Wq, hw + HW_Q, 262144);
  k_cvt<<<262144 / 2048, 256, 0, stream>>>(Wk, hw + HW_K, 262144);
  k_cvt<<<262144 / 2048, 256, 0, stream>>>(Wv, hw + HW_V, 262144);
  k_cvt<<<262144 / 2048, 256, 0, stream>>>(Wo, hw + HW_O, 262144);
  k_cvt<<<786432 / 2048, 256, 0, stream>>>(Wg1, hw + HW_G1, 786432);
  k_cvt<<<262144 / 2048, 256, 0, stream>>>(Wg2, hw + HW_G2, 262144);

  k0_nsw<<<1, 32, 0, stream>>>(sw, pres, out, ws);
  k1_pool<<<B_ * L_, 256, 0, stream>>>(sr, rmask, roles, pres, remb, ws, out);
  k2_attn<<<(B_ * L_) / 16, 256, sm2, stream>>>(sr, rmask, roles, pres, remb,
                                                ws, hw, out);
  k3_gate<<<(B_ * L_) / 16, 256, sm3, stream>>>(bg1, bg2, ln1g, ln1b, ng, nb,
                                                hw, out);
}
